// Attention_6038724018979
// MI455X (gfx1250) — matrix operand order repaired and checked by reading, not compiled, not run on hardware
//
#include <hip/hip_runtime.h>
#include <math.h>

typedef __attribute__((ext_vector_type(16))) __bf16 v16bf;
typedef __attribute__((ext_vector_type(8)))  float  v8f;
typedef __attribute__((ext_vector_type(4)))  float  v4f;

#define WMMA_BF16(A, Bm, C) \
  __builtin_amdgcn_wmma_f32_16x16x32_bf16(false, (A), false, (Bm), (short)0, (C), false, false)

// Problem constants (match reference setup_inputs)
constexpr int BB = 2, HH = 16, SS = 2048, DD = 64;
constexpr float SCALE   = 0.125f;   // 1/sqrt(64)
constexpr float NEG_INF = -1.0e9f;

constexpr int SC_STRIDE = 2052;     // 2048 + 4 pad floats (LDS bank de-conflict)
constexpr int OB_STRIDE = 68;       // 64 + 4 pad floats

// 16 contiguous fp32 -> bf16 fragment (B-matrix layout: elem e == K = lanehalf*16 + e)
__device__ __forceinline__ v16bf load16f_bf(const float* p) {
  const v4f* p4 = (const v4f*)p;
  v4f a = p4[0], b = p4[1], c = p4[2], d = p4[3];
  v16bf r;
  r[0]=(__bf16)a[0];  r[1]=(__bf16)a[1];  r[2]=(__bf16)a[2];  r[3]=(__bf16)a[3];
  r[4]=(__bf16)b[0];  r[5]=(__bf16)b[1];  r[6]=(__bf16)b[2];  r[7]=(__bf16)b[3];
  r[8]=(__bf16)c[0];  r[9]=(__bf16)c[1];  r[10]=(__bf16)c[2]; r[11]=(__bf16)c[3];
  r[12]=(__bf16)d[0]; r[13]=(__bf16)d[1]; r[14]=(__bf16)d[2]; r[15]=(__bf16)d[3];
  return r;
}

// A-matrix 16x32 bf16 fragment: elem e -> K = lanehalf*16 + e (callers pass hi = lo + 8).
// lo points at the 8-float run for K-group 0, hi at the run for K-group 1.
__device__ __forceinline__ v16bf load_a8x2(const float* lo, const float* hi) {
  const v4f* l4 = (const v4f*)lo;
  const v4f* h4 = (const v4f*)hi;
  v4f a = l4[0], b = l4[1], c = h4[0], d = h4[1];
  v16bf r;
  r[0]=(__bf16)a[0];  r[1]=(__bf16)a[1];  r[2]=(__bf16)a[2];  r[3]=(__bf16)a[3];
  r[4]=(__bf16)b[0];  r[5]=(__bf16)b[1];  r[6]=(__bf16)b[2];  r[7]=(__bf16)b[3];
  r[8]=(__bf16)c[0];  r[9]=(__bf16)c[1];  r[10]=(__bf16)c[2]; r[11]=(__bf16)c[3];
  r[12]=(__bf16)d[0]; r[13]=(__bf16)d[1]; r[14]=(__bf16)d[2]; r[15]=(__bf16)d[3];
  return r;
}

__global__ __launch_bounds__(256) void
attn_fused_kernel(const float* __restrict__ Q, const float* __restrict__ K,
                  const float* __restrict__ V, const int* __restrict__ mask,
                  float* __restrict__ out, float* __restrict__ pattn) {
  extern __shared__ char smem_raw[];
  float* sc  = (float*)smem_raw;            // [16][SC_STRIDE] scores -> probs (fp32)
  float* qld = sc + 16 * SC_STRIDE;         // [16][64] staged Q tile
  float* ob  = qld + 16 * DD;               // [16][OB_STRIDE] output combine buffer
  int*   qm  = (int*)(ob + 16 * OB_STRIDE); // [16] query-row mask

  const int qt = blockIdx.x, h = blockIdx.y, b = blockIdx.z;
  const int qbase = qt * 16;
  const int tid = threadIdx.x;
  const int lane = tid & 31, wv = tid >> 5;
  const int lhalf = lane >> 4, l16 = lane & 15;

  const size_t bh = (size_t)b * HH + h;
  const float* Qbh = Q + bh * (size_t)SS * DD;
  const float* Kbh = K + bh * (size_t)SS * DD;
  const float* Vbh = V + bh * (size_t)SS * DD;

  // ---- stage Q tile (16x64 fp32, coalesced) + query-row mask ----
  for (int i = tid; i < 16 * DD; i += 256)
    qld[i] = Qbh[(size_t)qbase * DD + i];
  if (tid < 16) qm[tid] = mask[(size_t)b * SS + qbase + tid];
  __syncthreads();

  // ---- build Q A-fragments (two 16x32 bf16 tiles covering D=64) ----
  v16bf qa0, qa1;
  {
    const float* qr = qld + l16 * DD + lhalf * 16;  // row = lane%16, K-offset = lanehalf*16
    qa0 = load_a8x2(qr,      qr + 8);               // dims  0..31
    qa1 = load_a8x2(qr + 32, qr + 40);              // dims 32..63
  }

  // Hoist the 8 query-row mask bits this lane needs (C rows M = r + lhalf*8)
  // into registers so the inner-loop mask is a branchless v_cndmask.
  int qmr[8];
#pragma unroll
  for (int r = 0; r < 8; ++r) qmr[r] = qm[r + lhalf * 8];

  // ---- phase 1: scores = scale * Q K^T, masked, into LDS ----
  const int kmbase = b * SS;
  for (int i = 0; i < 16; ++i) {
    const int kt = wv * 16 + i;                     // 128 key tiles / 8 waves
    const float* kr = Kbh + (size_t)(kt * 16 + l16) * DD + lhalf * 16;
    if (i < 15)
      __builtin_prefetch(Kbh + (size_t)((kt + 1) * 16 + l16) * DD, 0, 1);
    v16bf kb0 = load16f_bf(kr);        // dims  0..31 (B layout: N=lane%16)
    v16bf kb1 = load16f_bf(kr + 32);   // dims 32..63
    v8f acc = {};
    acc = WMMA_BF16(qa0, kb0, acc);
    acc = WMMA_BF16(qa1, kb1, acc);
    const int mk   = mask[kmbase + kt * 16 + l16];
    const int colg = kt * 16 + l16;
#pragma unroll
    for (int r = 0; r < 8; ++r) {
      const int row = r + lhalf * 8;                // C layout: M = vgpr + lanehalf*8
      const float v = (mk & qmr[r]) != 0 ? acc[r] * SCALE : NEG_INF;  // branchless select
      sc[row * SC_STRIDE + colg] = v;
    }
  }
  __syncthreads();

  // ---- phase 2: row softmax (2 rows per wave); write p_attn, keep P in LDS ----
  float* prowsBase = pattn + (bh * SS + qbase) * (size_t)SS;
#pragma unroll
  for (int rr = 0; rr < 2; ++rr) {
    const int row = wv * 2 + rr;
    float* srow = sc + row * SC_STRIDE;
    float rmax = -3.0e38f;
    for (int j = lane; j < SS; j += 32) rmax = fmaxf(rmax, srow[j]);
#pragma unroll
    for (int off = 16; off >= 1; off >>= 1)
      rmax = fmaxf(rmax, __shfl_xor(rmax, off, 32));
    float rsum = 0.f;
    for (int j = lane; j < SS; j += 32) {
      const float e = __expf(srow[j] - rmax);
      srow[j] = e;
      rsum += e;
    }
#pragma unroll
    for (int off = 16; off >= 1; off >>= 1)
      rsum += __shfl_xor(rsum, off, 32);
    const float inv = 1.0f / rsum;
    float* prow = prowsBase + (size_t)row * SS;
    for (int j = lane; j < SS; j += 32) {
      const float pv = srow[j] * inv;
      srow[j] = pv;                                  // keep normalized P for phase 3
      __builtin_nontemporal_store(pv, prow + j);     // stream p_attn (write-once, skip L2 hot set)
    }
  }
  __syncthreads();

  // ---- phase 3: out = P (16x2048) @ V (2048x64), bf16 WMMA, fp32 acc ----
  const int ntile = wv & 3;     // which 16-wide slice of D
  const int khalf = wv >> 2;    // which half of the 2048 K range
  v8f oacc = {};
  {
    const int rowA  = l16;
    const int koffA = lhalf * 16;
    for (int s = 0; s < 32; ++s) {
      const int kb = (khalf * 32 + s) * 32;         // base key index of this 32-chunk
      // A fragment from P (LDS, fp32 -> bf16)
      const float* pr = sc + rowA * SC_STRIDE + kb + koffA;
      v16bf a = load_a8x2(pr, pr + 8);
      // B fragment from V: key = kb + lanehalf*16 + e, d = ntile*16 + lane%16
      const float* vb = Vbh + (size_t)(kb + lhalf * 16) * DD + ntile * 16 + l16;
      v16bf bf;
#pragma unroll
      for (int e = 0; e < 16; ++e) bf[e] = (__bf16)vb[(size_t)e * DD];
      oacc = WMMA_BF16(a, bf, oacc);
    }
  }
  // combine the two K-halves via LDS
  if (khalf == 0) {
#pragma unroll
    for (int r = 0; r < 8; ++r)
      ob[(r + lhalf * 8) * OB_STRIDE + ntile * 16 + l16] = oacc[r];
  }
  __syncthreads();
  if (khalf == 1) {
#pragma unroll
    for (int r = 0; r < 8; ++r)
      ob[(r + lhalf * 8) * OB_STRIDE + ntile * 16 + l16] += oacc[r];
  }
  __syncthreads();

  // ---- write out tile (16x64) as v4f, coalesced, non-temporal ----
  {
    const int row = tid >> 4;
    const int d4  = (tid & 15) * 4;
    const v4f o = *(const v4f*)(ob + row * OB_STRIDE + d4);
    __builtin_nontemporal_store(o, (v4f*)(out + (bh * SS + qbase + row) * (size_t)DD + d4));
  }
}

extern "C" void kernel_launch(void* const* d_in, const int* in_sizes, int n_in,
                              void* d_out, int out_size, void* d_ws, size_t ws_size,
                              hipStream_t stream) {
  const float* Q    = (const float*)d_in[0];
  const float* K    = (const float*)d_in[1];
  const float* V    = (const float*)d_in[2];
  const int*   mask = (const int*)d_in[3];

  float* out   = (float*)d_out;
  float* pattn = out + (size_t)BB * HH * SS * DD;   // outputs concatenated: (out, p_attn)

  const size_t shmem =
      (size_t)(16 * SC_STRIDE + 16 * DD + 16 * OB_STRIDE) * sizeof(float) +
      16 * sizeof(int);

  // Allow >64KB dynamic LDS (WGP has 320KB). Not a stream op: graph-capture safe.
  (void)hipFuncSetAttribute((const void*)attn_fused_kernel,
                            hipFuncAttributeMaxDynamicSharedMemorySize, (int)shmem);

  dim3 grid(SS / 16, HH, BB);   // 128 x 16 x 2 = 4096 workgroups
  dim3 block(256);              // 8 wave32s
  attn_fused_kernel<<<grid, block, shmem, stream>>>(Q, K, V, mask, out, pattn);
}
